// HybridQLSTM_65481071396414
// MI455X (gfx1250) — matrix operand order repaired and checked by reading, not compiled, not run on hardware
//
#include <hip/hip_runtime.h>
#include <hip/hip_bf16.h>
#include <math.h>

typedef __attribute__((ext_vector_type(16))) _Float16 v16h;
typedef __attribute__((ext_vector_type(8)))  float    v8f;

#define T_STEPS 128
#define BATCH   512
#define D_IN    128
#define NG      32      // 4 gates * 8 hidden
#define NH      8

// workspace layout (bytes); ws base from hipMalloc -> 256B aligned
#define XPROJ_OFF  0ULL                                   // 65536*32 f32 = 8 MB
#define BPACK_OFF  ((unsigned long long)T_STEPS*BATCH*NG*4) // 4096 f16 = 8 KB
#define WH_OFF     (BPACK_OFF + 8192ULL)                  // 256 f32 = 1 KB
#define BIAS_OFF   (WH_OFF + 1024ULL)                     // 32 f32
#define THETA_OFF  (BIAS_OFF + 128ULL)                    // 32 f32

// ---------------------------------------------------------------------------
// Prep: pack B (x-part of weights) into WMMA B-register layout (f16), extract
// the recurrent weights Wh (rows 128..135), and fuse bias+theta vectors.
// B layout per CDNA5 ISA: element e of lane l holds B[K = (e<8 ? 8*(l>>4)+e : 16+8*(l>>4)+(e-8))][N = l&15].
// ---------------------------------------------------------------------------
__global__ void qlstm_prep(const float* __restrict__ Wf, const float* __restrict__ Wi,
                           const float* __restrict__ Wu, const float* __restrict__ Wo,
                           const float* __restrict__ bf, const float* __restrict__ bi,
                           const float* __restrict__ bu, const float* __restrict__ bo,
                           const float* __restrict__ tf, const float* __restrict__ ti,
                           const float* __restrict__ tu, const float* __restrict__ to_,
                           _Float16* __restrict__ bpack, float* __restrict__ wh,
                           float* __restrict__ bias, float* __restrict__ theta) {
    const float* Ws[4] = {Wf, Wi, Wu, Wo};
    const float* Bs[4] = {bf, bi, bu, bo};
    const float* Ts[4] = {tf, ti, tu, to_};
    int tid = threadIdx.x;  // 256 threads, 1 block
    // bpack: index i = ((s*2+nt)*32 + lane)*16 + e
    for (int i = tid; i < 4096; i += 256) {
        int e  = i & 15;
        int l  = (i >> 4) & 31;
        int nt = (i >> 9) & 1;
        int s  = i >> 10;
        int k  = s * 32 + (e < 8 ? 8 * (l >> 4) + e : 16 + 8 * (l >> 4) + (e - 8));   // K in 0..127
        int n  = nt * 16 + (l & 15);           // N in 0..31
        bpack[i] = (_Float16)Ws[n >> 3][k * NH + (n & 7)];
    }
    // Wh: [k=0..7][n=0..31] from rows 128..135 of each gate's W
    {
        int k = tid >> 5, n = tid & 31;
        wh[tid] = Ws[n >> 3][(D_IN + k) * NH + (n & 7)];
    }
    if (tid < NG) {
        bias[tid]  = Bs[tid >> 3][tid & 7];
        theta[tid] = Ts[tid >> 3][tid & 7];
    }
}

// ---------------------------------------------------------------------------
// GEMM: xproj[t*B+b, 0..31] = x[t,b,:] @ Wx[128,32]  (f16 WMMA, f32 acc)
// One wave per 16-row tile; 4 k-steps x 2 n-tiles = 8 v_wmma per wave.
// ---------------------------------------------------------------------------
__global__ void qlstm_xproj(const float* __restrict__ x,
                            const _Float16* __restrict__ bpack,
                            float* __restrict__ xproj) {
    int gid  = blockIdx.x * blockDim.x + threadIdx.x;
    int wave = gid >> 5;                 // 0..4095
    int lane = threadIdx.x & 31;
    int m    = lane & 15;
    int g    = lane >> 4;                // lane-group: K-offset selector
    const float* xrow = x + (size_t)(wave * 16 + m) * D_IN;
    const v16h*  bp   = (const v16h*)bpack;

    v8f acc0 = {};
    v8f acc1 = {};
#pragma unroll
    for (int s = 0; s < 4; ++s) {
        // A tile, ISA 16-bit A layout: e<8 -> K = s*32 + 8g + e ; e>=8 -> K = s*32+16+8g+(e-8)
        const float4* p0 = (const float4*)(xrow + s * 32 + 8 * g);
        const float4* p1 = (const float4*)(xrow + s * 32 + 16 + 8 * g);
        float4 f0 = p0[0], f1 = p0[1];
        float4 f2 = p1[0], f3 = p1[1];
        v16h a;
        a[0]  = (_Float16)f0.x; a[1]  = (_Float16)f0.y; a[2]  = (_Float16)f0.z; a[3]  = (_Float16)f0.w;
        a[4]  = (_Float16)f1.x; a[5]  = (_Float16)f1.y; a[6]  = (_Float16)f1.z; a[7]  = (_Float16)f1.w;
        a[8]  = (_Float16)f2.x; a[9]  = (_Float16)f2.y; a[10] = (_Float16)f2.z; a[11] = (_Float16)f2.w;
        a[12] = (_Float16)f3.x; a[13] = (_Float16)f3.y; a[14] = (_Float16)f3.z; a[15] = (_Float16)f3.w;

        v16h b0 = bp[(s * 2 + 0) * 32 + lane];
        v16h b1 = bp[(s * 2 + 1) * 32 + lane];

        acc0 = __builtin_amdgcn_wmma_f32_16x16x32_f16(false, a, false, b0, (short)0, acc0, false, false);
        acc1 = __builtin_amdgcn_wmma_f32_16x16x32_f16(false, a, false, b1, (short)0, acc1, false, false);
    }
    // C/D layout: VGPR r, lane l -> row M = r + 8*(l>>4), col N = l&15
#pragma unroll
    for (int r = 0; r < 8; ++r) {
        size_t row = (size_t)(wave * 16 + r + 8 * g);
        xproj[row * NG + (lane & 15)]      = acc0[r];
        xproj[row * NG + 16 + (lane & 15)] = acc1[r];
    }
}

// ---------------------------------------------------------------------------
// Recurrence: one thread per batch element, 128 sequential steps.
// q_k = prefix products of cos(angle); gates f,i,o = sigmoid, u = tanh.
// ---------------------------------------------------------------------------
__device__ __forceinline__ float sigm(float v) { return 1.0f / (1.0f + expf(-v)); }

__global__ void qlstm_recur(const float* __restrict__ xproj,
                            const float* __restrict__ wh,
                            const float* __restrict__ bias,
                            const float* __restrict__ theta,
                            float* __restrict__ out) {
    __shared__ float sWh[NH * NG];
    __shared__ float sB[NG];
    __shared__ float sTh[NG];
    int tid = threadIdx.x;  // 32 threads per block, 16 blocks
    for (int i = tid; i < NH * NG; i += 32) sWh[i] = wh[i];
    sB[tid & 31]  = bias[tid & 31];
    sTh[tid & 31] = theta[tid & 31];
    __syncthreads();

    int b = blockIdx.x * 32 + tid;  // 0..511
    float hx[NH], cx[NH];
#pragma unroll
    for (int j = 0; j < NH; ++j) { hx[j] = 0.0f; cx[j] = 0.0f; }

    float* outs = out;                                   // (128, 512, 8)
    float* hout = out + (size_t)T_STEPS * BATCH * NH;    // (512, 8)
    float* cout = hout + (size_t)BATCH * NH;             // (512, 8)

    for (int t = 0; t < T_STEPS; ++t) {
        float a[NG];
        const float4* xp = (const float4*)(xproj + ((size_t)t * BATCH + b) * NG);
#pragma unroll
        for (int q = 0; q < 8; ++q) {
            float4 v = xp[q];
            a[q * 4 + 0] = v.x; a[q * 4 + 1] = v.y; a[q * 4 + 2] = v.z; a[q * 4 + 3] = v.w;
        }
#pragma unroll
        for (int j = 0; j < NG; ++j) a[j] += sB[j] + sTh[j];
        // recurrent projection: a += hx @ Wh
#pragma unroll
        for (int k = 0; k < NH; ++k) {
            float h = hx[k];
#pragma unroll
            for (int j = 0; j < NG; ++j) a[j] = fmaf(h, sWh[k * NG + j], a[j]);
        }
        float cp[NG];
#pragma unroll
        for (int j = 0; j < NG; ++j) cp[j] = cosf(a[j]);

        // quantum layer (analytic): q[0]=prod_{1..7} cos, q[k>=1]=prod_{0..k} cos
        float qv[NG];
#pragma unroll
        for (int gi = 0; gi < 4; ++gi) {
            const float* c = cp + gi * NH;
            float pref = c[0];
#pragma unroll
            for (int k = 1; k < NH; ++k) { pref *= c[k]; qv[gi * NH + k] = pref; }
            float p17 = c[1];
#pragma unroll
            for (int k = 2; k < NH; ++k) p17 *= c[k];
            qv[gi * NH + 0] = p17;
        }
        float* orow = outs + ((size_t)t * BATCH + b) * NH;
#pragma unroll
        for (int j = 0; j < NH; ++j) {
            float fg = sigm(qv[0 * NH + j]);
            float ig = sigm(qv[1 * NH + j]);
            float ug = tanhf(qv[2 * NH + j]);
            float og = sigm(qv[3 * NH + j]);
            cx[j] = fg * cx[j] + ig * ug;
            hx[j] = og * tanhf(cx[j]);
            orow[j] = hx[j];
        }
    }
#pragma unroll
    for (int j = 0; j < NH; ++j) {
        hout[(size_t)b * NH + j] = hx[j];
        cout[(size_t)b * NH + j] = cx[j];
    }
}

extern "C" void kernel_launch(void* const* d_in, const int* in_sizes, int n_in,
                              void* d_out, int out_size, void* d_ws, size_t ws_size,
                              hipStream_t stream) {
    const float* x  = (const float*)d_in[0];
    const float* Wf = (const float*)d_in[1];
    const float* bf = (const float*)d_in[2];
    const float* tf = (const float*)d_in[3];
    const float* Wi = (const float*)d_in[4];
    const float* bi = (const float*)d_in[5];
    const float* ti = (const float*)d_in[6];
    const float* Wu = (const float*)d_in[7];
    const float* bu = (const float*)d_in[8];
    const float* tu = (const float*)d_in[9];
    const float* Wo = (const float*)d_in[10];
    const float* bo = (const float*)d_in[11];
    const float* to_ = (const float*)d_in[12];

    char* ws = (char*)d_ws;
    float*    xproj = (float*)(ws + XPROJ_OFF);
    _Float16* bpack = (_Float16*)(ws + BPACK_OFF);
    float*    wh    = (float*)(ws + WH_OFF);
    float*    bias  = (float*)(ws + BIAS_OFF);
    float*    theta = (float*)(ws + THETA_OFF);

    qlstm_prep<<<1, 256, 0, stream>>>(Wf, Wi, Wu, Wo, bf, bi, bu, bo, tf, ti, tu, to_,
                                      bpack, wh, bias, theta);
    // 4096 tiles * 32 lanes = 131072 threads
    qlstm_xproj<<<512, 256, 0, stream>>>(x, bpack, xproj);
    // 512 batch elements, 1 wave per block spread across WGPs
    qlstm_recur<<<16, 32, 0, stream>>>(xproj, wh, bias, theta, (float*)d_out);
}
